// Transformer_83906481094968
// MI455X (gfx1250) — matrix operand order repaired and checked by reading, not compiled, not run on hardware
//
#include <hip/hip_runtime.h>

// ---------------------------------------------------------------------------
// MI455X / gfx1250. bf16 WMMA (v_wmma_f32_16x16x32_bf16, f32 accum) for all
// GEMMs + attention. Weights/activations pre-converted to bf16 once; tanh is
// fused into GEMM epilogues. att1/attx evaluated only at the last sequence
// position. GEMM K-loop is software-pipelined: next tile's global loads are
// issued before the current tile's WMMAs so memory latency overlaps compute.
// Workspace: ~308 MB.
// ---------------------------------------------------------------------------

typedef __bf16 bf16;
typedef __attribute__((ext_vector_type(16))) bf16  v16bf;
typedef __attribute__((ext_vector_type(8)))  bf16  bf16x8;
typedef __attribute__((ext_vector_type(2)))  bf16  bf16x2;
typedef __attribute__((ext_vector_type(8)))  float v8f;

#define B_   256
#define L_   128
#define D_   512
#define H_   8
#define DH_  64
#define NBLD (256L * 128L * 512L)

__device__ __forceinline__ float fast_tanh(float x) {
  float a = fabsf(x);
  float e = __expf(-2.f * a);          // stable for large |x|
  float t = (1.f - e) / (1.f + e);
  return copysignf(t, x);
}

// A-matrix (16x32 bf16) per-lane K offset for element-pair v (contiguous K, matches B).
__device__ __forceinline__ int frag_koff(int v, int hi) {
  return hi * 16 + 2 * v;
}

__device__ __forceinline__ v8f wmma_bf16(v16bf a, v16bf b, v8f c) {
  return __builtin_amdgcn_wmma_f32_16x16x32_bf16(false, a, false, b, (short)0,
                                                 c, false, false);
}

// ---------------------------------------------------------------------------
// GEMM: acc = A_bf16[M,K] @ W_bf16[K,N]; v = (acc+bias)*alpha; relu?; +R?;
// outputs: C (fp32, optional) and/or Cb (bf16, optional, tanh-fused optional).
// WG tile 128x128, 8 waves (4Mx2N), wave tile 32x64, K-step 32 via LDS.
// Software-pipelined staging (regs hold tile k+1 while tile k computes).
// Requires M%128==0, N%128==0, K%32==0 (true for all calls here).
// ---------------------------------------------------------------------------
__global__ __launch_bounds__(256)
void gemm_bf16_wmma(const bf16* __restrict__ A, long lda,
                    const bf16* __restrict__ W, long ldw,
                    const float* __restrict__ bias, float alpha, int relu,
                    const float* __restrict__ R, long ldr,
                    float* __restrict__ C, bf16* __restrict__ Cb, int tanh_out,
                    long ldc, int M, int N, int K) {
  __shared__ bf16 As[128][40];   // [m][k], stride 40 keeps 16B chunk alignment
  __shared__ bf16 Bs[128][40];   // [n][k] (k-contiguous per output column)

  const int t = threadIdx.x;
  const int wid = t >> 5, lane = t & 31, hi = lane >> 4, ln = lane & 15;
  const int m0 = blockIdx.y * 128, n0 = blockIdx.x * 128;
  const int wm = (wid & 3) * 32, wn = (wid >> 2) * 64;

  // per-thread staging coordinates (hoisted out of the K-loop)
  const int arow = t >> 2, ac = 8 * (t & 3);      // A: 2 chunks (rows r, r+64)
  const int bk = t >> 4, bn = 8 * (t & 15);       // B: 2 chunks (k, k+16)
  const bf16* pA0 = A + (long)(m0 + arow) * lda + ac;
  const bf16* pA1 = pA0 + 64 * lda;
  const bf16* pW0 = W + (long)bk * ldw + n0 + bn;
  const bf16* pW1 = pW0 + 16 * ldw;
  bf16* sA0 = &As[arow][ac];
  bf16* sA1 = &As[64 + arow][ac];

  v8f acc[2][4];
  const v8f vzero = {0.f, 0.f, 0.f, 0.f, 0.f, 0.f, 0.f, 0.f};
#pragma unroll
  for (int i = 0; i < 2; ++i)
#pragma unroll
    for (int j = 0; j < 4; ++j) acc[i][j] = vzero;

  // prime the pipeline with tile 0
  bf16x8 ra0 = *(const bf16x8*)pA0, ra1 = *(const bf16x8*)pA1;
  bf16x8 rb0 = *(const bf16x8*)pW0, rb1 = *(const bf16x8*)pW1;

  const int nk = K >> 5;
  for (int ks = 0; ks < nk; ++ks) {
    // store staged registers to LDS
    *(bf16x8*)sA0 = ra0;
    *(bf16x8*)sA1 = ra1;
#pragma unroll
    for (int e = 0; e < 8; ++e) Bs[bn + e][bk] = rb0[e];
#pragma unroll
    for (int e = 0; e < 8; ++e) Bs[bn + e][16 + bk] = rb1[e];
    __syncthreads();

    // issue next tile's global loads; they overlap the WMMAs below
    if (ks + 1 < nk) {
      pA0 += 32; pA1 += 32;
      pW0 += 32 * ldw; pW1 += 32 * ldw;
      ra0 = *(const bf16x8*)pA0; ra1 = *(const bf16x8*)pA1;
      rb0 = *(const bf16x8*)pW0; rb1 = *(const bf16x8*)pW1;
    }

    v16bf af[2];
#pragma unroll
    for (int i = 0; i < 2; ++i) {
      int row = wm + i * 16 + ln;
#pragma unroll
      for (int v = 0; v < 8; ++v) {
        bf16x2 p = *(const bf16x2*)&As[row][frag_koff(v, hi)];
        af[i][2 * v] = p.x; af[i][2 * v + 1] = p.y;
      }
    }
#pragma unroll
    for (int j = 0; j < 4; ++j) {
      v16bf bfr;
      int col = wn + j * 16 + ln;
#pragma unroll
      for (int v = 0; v < 8; ++v) {         // B lanes0-15:K0-15, 16-31:K16-31
        bf16x2 p = *(const bf16x2*)&Bs[col][hi * 16 + 2 * v];
        bfr[2 * v] = p.x; bfr[2 * v + 1] = p.y;
      }
#pragma unroll
      for (int i = 0; i < 2; ++i) acc[i][j] = wmma_bf16(af[i], bfr, acc[i][j]);
    }
    __syncthreads();
  }

#pragma unroll
  for (int i = 0; i < 2; ++i)
#pragma unroll
    for (int j = 0; j < 4; ++j) {
      int col = n0 + wn + j * 16 + ln;
      float bb = bias ? bias[col] : 0.f;
#pragma unroll
      for (int r = 0; r < 8; ++r) {
        int row = m0 + wm + i * 16 + r + hi * 8;   // C layout: N=lane, M=r+8*hi
        float v = (acc[i][j][r] + bb) * alpha;
        if (relu) v = fmaxf(v, 0.f);
        if (R) v += R[(long)row * ldr + col];
        if (C)  C[(long)row * ldc + col] = v;
        if (Cb) Cb[(long)row * ldc + col] = (bf16)(tanh_out ? fast_tanh(v) : v);
      }
    }
}

// ---------------------------------------------------------------------------
// Fused attention per (b,h): S=QK^T (WMMA), softmax in LDS, O=P*V (WMMA).
// Q/K/V/O all bf16 [B,L,H*DH]; Q pre-scaled/biased. LDS ~120 KB (<320 KB).
// ---------------------------------------------------------------------------
__global__ __launch_bounds__(256)
void attn_full_kernel(const bf16* __restrict__ Qg, const bf16* __restrict__ Kg,
                      const bf16* __restrict__ Vg, bf16* __restrict__ Og) {
  const int b = blockIdx.x >> 3, h = blockIdx.x & 7;
  const int t = threadIdx.x;
  const int wid = t >> 5, lane = t & 31, hi = lane >> 4, ln = lane & 15;

  __shared__ bf16  Qs[128][72];
  __shared__ bf16  Ks[128][72];
  __shared__ bf16  Vt[64][136];     // V transposed: Vt[d][l]
  __shared__ float Ss[128][129];    // logits -> probabilities (fp32)

  const long base = ((long)b * L_) * D_ + (long)h * DH_;
  {                                           // batched register staging
    bf16x8 rq[4], rk[4], rv[4];
#pragma unroll
    for (int i = 0; i < 4; ++i) {             // 1024 x (8-elem chunks)
      int idx = i * 256 + t;
      long g = base + (long)(idx >> 3) * D_ + 8 * (idx & 7);
      rq[i] = *(const bf16x8*)&Qg[g];
      rk[i] = *(const bf16x8*)&Kg[g];
      rv[i] = *(const bf16x8*)&Vg[g];
    }
#pragma unroll
    for (int i = 0; i < 4; ++i) {
      int idx = i * 256 + t;
      int r = idx >> 3, c8 = 8 * (idx & 7);
      *(bf16x8*)&Qs[r][c8] = rq[i];
      *(bf16x8*)&Ks[r][c8] = rk[i];
#pragma unroll
      for (int e = 0; e < 8; ++e) Vt[c8 + e][r] = rv[i][e];
    }
  }
  __syncthreads();

  { // S = Q K^T : wave owns 16 query rows x 8 key tiles, K-dim 64 = 2 steps
    v8f accS[8];
    const v8f vzero = {0.f, 0.f, 0.f, 0.f, 0.f, 0.f, 0.f, 0.f};
#pragma unroll
    for (int j = 0; j < 8; ++j) accS[j] = vzero;
#pragma unroll
    for (int ks = 0; ks < 2; ++ks) {
      v16bf a;
      int row = wid * 16 + ln;
#pragma unroll
      for (int v = 0; v < 8; ++v) {
        bf16x2 p = *(const bf16x2*)&Qs[row][ks * 32 + frag_koff(v, hi)];
        a[2 * v] = p.x; a[2 * v + 1] = p.y;
      }
#pragma unroll
      for (int jt = 0; jt < 8; ++jt) {
        v16bf bfr;
        int col = jt * 16 + ln;
#pragma unroll
        for (int v = 0; v < 8; ++v) {
          bf16x2 p = *(const bf16x2*)&Ks[col][ks * 32 + hi * 16 + 2 * v];
          bfr[2 * v] = p.x; bfr[2 * v + 1] = p.y;
        }
        accS[jt] = wmma_bf16(a, bfr, accS[jt]);
      }
    }
#pragma unroll
    for (int jt = 0; jt < 8; ++jt)
#pragma unroll
      for (int r = 0; r < 8; ++r)
        Ss[wid * 16 + r + hi * 8][jt * 16 + ln] = accS[jt][r];
  }
  __syncthreads();

  if (t < 128) {                              // row softmax (max-shifted)
    float m = -3.4e38f;
    for (int j = 0; j < 128; ++j) m = fmaxf(m, Ss[t][j]);
    float ssum = 0.f;
    for (int j = 0; j < 128; ++j) { float e = __expf(Ss[t][j] - m); Ss[t][j] = e; ssum += e; }
    float inv = 1.f / ssum;
    for (int j = 0; j < 128; ++j) Ss[t][j] *= inv;
  }
  __syncthreads();

  { // O = P * V : wave owns 16 rows x 4 d-tiles, K(j)-dim 128 = 4 steps
    v8f accO[4];
    const v8f vzero = {0.f, 0.f, 0.f, 0.f, 0.f, 0.f, 0.f, 0.f};
#pragma unroll
    for (int d = 0; d < 4; ++d) accO[d] = vzero;
#pragma unroll
    for (int ks = 0; ks < 4; ++ks) {
      v16bf a;
      int row = wid * 16 + ln;
#pragma unroll
      for (int v = 0; v < 8; ++v) {           // P fp32 -> bf16 on the fly
        int j0 = ks * 32 + frag_koff(v, hi);
        a[2 * v]     = (bf16)Ss[row][j0];
        a[2 * v + 1] = (bf16)Ss[row][j0 + 1];
      }
#pragma unroll
      for (int dt = 0; dt < 4; ++dt) {
        v16bf bfr;
        int col = dt * 16 + ln;
#pragma unroll
        for (int v = 0; v < 8; ++v) {
          bf16x2 p = *(const bf16x2*)&Vt[col][ks * 32 + hi * 16 + 2 * v];
          bfr[2 * v] = p.x; bfr[2 * v + 1] = p.y;
        }
        accO[dt] = wmma_bf16(a, bfr, accO[dt]);
      }
    }
#pragma unroll
    for (int dt = 0; dt < 4; ++dt)
#pragma unroll
      for (int r = 0; r < 8; ++r) {
        int row = wid * 16 + r + hi * 8;
        Og[base + (long)row * D_ + dt * 16 + ln] = (bf16)accO[dt][r];
      }
  }
}

// Single-query attention at position L-1 (att1 / attx): per-(b,h) block.
__global__ __launch_bounds__(128)
void attn_single_kernel(const float* __restrict__ Q,   // [B,512] scaled+biased
                        const bf16* __restrict__ Kf,   // [B,L,512] bf16
                        const bf16* __restrict__ Vf,   // [B,L,512] bf16
                        bf16* __restrict__ O) {        // [B,512] bf16
  const int b = blockIdx.x >> 3, h = blockIdx.x & 7;
  const int t = threadIdx.x;                 // key index
  __shared__ float sm[128];
  const float* q  = Q  + (long)b * 512 + h * 64;
  const bf16*  kr = Kf + ((long)b * L_ + t) * 512 + h * 64;
  float s = 0.f;
#pragma unroll
  for (int d = 0; d < 64; ++d) s += q[d] * (float)kr[d];
  sm[t] = s;
  __syncthreads();
  float m = -3.4e38f;
  for (int j = 0; j < 128; ++j) m = fmaxf(m, sm[j]);
  __syncthreads();
  sm[t] = __expf(s - m);
  __syncthreads();
  if (t < 64) {
    float tot = 0.f, acc = 0.f;
    for (int j = 0; j < 128; ++j) tot += sm[j];
    for (int l = 0; l < 128; ++l)
      acc += sm[l] * (float)Vf[((long)b * L_ + l) * 512 + h * 64 + t];
    O[(long)b * 512 + h * 64 + t] = (bf16)(acc / tot);
  }
}

// Build: masked carry + slot0<-att_in; emit rolled new_carry, a2 fp32,
// tanh(a1)/tanh(a2) bf16, and a1's last row (fp32 residual).
__global__ __launch_bounds__(256)
void build_kernel(const float* __restrict__ carry, const int* __restrict__ resets,
                  const float* __restrict__ attin, float* __restrict__ newcarry,
                  float* __restrict__ A2f, bf16* __restrict__ T1,
                  bf16* __restrict__ T2, float* __restrict__ A1last) {
  const int b = blockIdx.x >> 7, l = blockIdx.x & 127;
  const int rz = resets[b];
  const float kC = -0.0179889460390160f;     // -ln(10000)/512
#pragma unroll
  for (int i = 0; i < 4; ++i) {
    int c = i * 256 + threadIdx.x;           // 0..1023
    float v = (l == 0) ? attin[(long)b * 1024 + c]
                       : (rz ? 0.f : carry[((long)b * L_ + l) * 1024 + c]);
    newcarry[((long)b * L_ + ((l + 1) & (L_ - 1))) * 1024 + c] = v;   // roll
    int d = c & 511;
    float ang = (float)l * __expf((float)(d & ~1) * kC);
    float pe = (d & 1) ? __cosf(ang) : __sinf(ang);
    float av = v + pe;
    long o = ((long)b * L_ + l) * 512 + d;
    if (c < 512) {
      T1[o] = (bf16)fast_tanh(av);
      if (l == L_ - 1) A1last[(long)b * 512 + d] = av;
    } else {
      A2f[o] = av;
      T2[o] = (bf16)fast_tanh(av);
    }
  }
}

__global__ __launch_bounds__(256)
void mask_h_kernel(const float* __restrict__ h, const int* __restrict__ resets,
                   float* __restrict__ of, bf16* __restrict__ ob) {
  int i = blockIdx.x * 256 + threadIdx.x;    // B*D
  float v = resets[i >> 9] ? 0.f : h[i];
  of[i] = v; ob[i] = (bf16)v;
}

__global__ __launch_bounds__(256)
void cvt_bf16_kernel(const float* __restrict__ s, bf16* __restrict__ d, int n) {
  int i = blockIdx.x * 256 + threadIdx.x;
  if (i < n) d[i] = (bf16)s[i];
}

__global__ __launch_bounds__(256)
void gru_final_kernel(const float* __restrict__ GI, const float* __restrict__ GH,
                      const float* __restrict__ Hm, const float* __restrict__ AX,
                      const float* __restrict__ bhn,
                      float* __restrict__ out_h, float* __restrict__ out_y) {
  int i = blockIdx.x * 256 + threadIdx.x;    // B*D
  int b = i >> 9, d = i & 511;
  long g = (long)b * 1536;
  float r = 1.f / (1.f + __expf(-(GI[g + d]        + GH[g + d])));
  float z = 1.f / (1.f + __expf(-(GI[g + 512 + d]  + GH[g + 512 + d])));
  float n = fast_tanh(GI[g + 1024 + d] + r * (GH[g + 1024 + d] + bhn[d]));
  float hv = (1.f - z) * n + z * Hm[i];
  out_h[i] = hv;
  out_y[i] = hv + AX[i];
}

// ---------------------------------------------------------------------------
extern "C" void kernel_launch(void* const* d_in, const int* in_sizes, int n_in,
                              void* d_out, int out_size, void* d_ws, size_t ws_size,
                              hipStream_t stream) {
  const float* carry  = (const float*)d_in[0];
  const float* rnn    = (const float*)d_in[1];
  const float* in1    = (const float*)d_in[2];
  const float* in2    = (const float*)d_in[3];
  const int*   resets = (const int*)d_in[4];
  const float* dense1 = (const float*)d_in[5];
  const float* dense2 = (const float*)d_in[6];
  // P[0..7]=att1{wq,bq,wk,bk,wv,bv,wo,bo}, P[8..15]=att2, P[16..23]=attx,
  // P[24,25]=ff1{w,b}, P[26,27]=ff2{w,b}, P[28..31]=gru{wi,bi,wh,bhn}
  const float* P[32];
  for (int i = 0; i < 32 && (7 + i) < n_in; ++i) P[i] = (const float*)d_in[7 + i];

  // ---- workspace carve ----
  float* wsf = (float*)d_ws;
  float* FA2 = wsf;                 // a2 fp32 (residual for y2)
  float* FY2 = wsf + NBLD;          // y2 fp32 (residual for ff2)
  float* sp  = wsf + 2 * NBLD;
  float* S_ATTIN = sp; sp += 256 * 1024;
  float* S_H   = sp; sp += 256 * 512;
  float* S_A1L = sp; sp += 256 * 512;   // a1[:, L-1, :]
  float* S_Q1  = sp; sp += 256 * 512;
  float* S_Y1  = sp; sp += 256 * 512;
  float* S_T   = sp; sp += 256 * 512;   // o1 at L-1 (fp32 residual)
  float* S_QX  = sp; sp += 256 * 512;
  float* S_AX  = sp; sp += 256 * 512;   // attx out at L-1
  float* S_GI  = sp; sp += 256 * 1536;
  float* S_GH  = sp; sp += 256 * 1536;

  bf16* bp = (bf16*)sp;
  bf16* T1 = bp;              bp += NBLD;   // tanh(a1); later attention O
  bf16* T2 = bp;              bp += NBLD;   // tanh(a2); later tanh(y2)
  bf16* G0 = bp;              bp += NBLD;   // K buffers
  bf16* G1 = bp;              bp += NBLD;   // V buffers
  bf16* G2 = bp;              bp += NBLD;   // att2 Q; later tanh(o2)
  bf16* S_HB   = bp; bp += 256 * 512;
  bf16* S_O1B  = bp; bp += 256 * 512;
  bf16* S_Y1TB = bp; bp += 256 * 512;
  bf16* S_TB   = bp; bp += 256 * 512;   // tanh(o1) at L-1
  bf16* S_OXB  = bp; bp += 256 * 512;
  bf16* S_AXTB = bp; bp += 256 * 512;   // tanh(attx out)
  bf16* IN1B   = bp; bp += 256 * 512;
  bf16* IN2B   = bp; bp += 256 * 512;
  // bf16 weights
  const float* wsrc[18] = {dense1, dense2, P[0], P[2], P[4], P[6],
                           P[8], P[10], P[12], P[14], P[16], P[18], P[20], P[22],
                           P[24], P[26], P[28], P[30]};
  int wn[18]; for (int i = 0; i < 16; ++i) wn[i] = 512 * 512;
  wn[16] = 512 * 1536; wn[17] = 512 * 1536;
  bf16* wdst[18];
  for (int i = 0; i < 18; ++i) { wdst[i] = bp; bp += wn[i]; }
  bf16 *WD1 = wdst[0], *WD2 = wdst[1];
  bf16 *WQ1 = wdst[2], *WK1 = wdst[3], *WV1 = wdst[4], *WO1 = wdst[5];
  bf16 *WQ2 = wdst[6], *WK2 = wdst[7], *WV2 = wdst[8], *WO2 = wdst[9];
  bf16 *WQX = wdst[10], *WKX = wdst[11], *WVX = wdst[12], *WOX = wdst[13];
  bf16 *WF1 = wdst[14], *WF2 = wdst[15], *WI = wdst[16], *WH = wdst[17];

  float* out       = (float*)d_out;
  float* out_carry = out;
  float* out_h     = out + (long)B_ * L_ * 2 * D_;
  float* out_y     = out_h + (long)B_ * D_;

  const dim3 blk(256);
  for (int i = 0; i < 18; ++i)
    cvt_bf16_kernel<<<dim3(wn[i] / 256), blk, 0, stream>>>(wsrc[i], wdst[i], wn[i]);
  cvt_bf16_kernel<<<dim3(512), blk, 0, stream>>>(in1, IN1B, 256 * 512);
  cvt_bf16_kernel<<<dim3(512), blk, 0, stream>>>(in2, IN2B, 256 * 512);

  auto gemm = [&](const bf16* A, long lda, const bf16* W, long ldw,
                  const float* bias, float alpha, int relu,
                  const float* R, long ldr,
                  float* C, bf16* Cb, int tanh_out, long ldc,
                  int M, int N, int K) {
    dim3 g(N / 128, M / 128);
    gemm_bf16_wmma<<<g, blk, 0, stream>>>(A, lda, W, ldw, bias, alpha, relu,
                                          R, ldr, C, Cb, tanh_out, ldc, M, N, K);
  };
  const float sc = 0.125f;          // 1/sqrt(DH)
  const long  LD = (long)L_ * D_;   // 65536
  const int   ML = B_ * L_;         // 32768

  // input projections -> att_in [B, 2D] (fp32)
  gemm(IN1B, 512, WD1, 512, nullptr, 1.f, 0, nullptr, 0, S_ATTIN,       nullptr, 0, 1024, B_, 512, 512);
  gemm(IN2B, 512, WD2, 512, nullptr, 1.f, 0, nullptr, 0, S_ATTIN + 512, nullptr, 0, 1024, B_, 512, 512);
  build_kernel<<<dim3(B_ * L_), blk, 0, stream>>>(carry, resets, S_ATTIN,
                                                  out_carry, FA2, T1, T2, S_A1L);
  mask_h_kernel<<<dim3(B_ * D_ / 256), blk, 0, stream>>>(rnn, resets, S_H, S_HB);

  // ---- att1 (only L-1 needed downstream) ----
  gemm(T1, 512, WK1, 512, P[3], 1.f, 0, nullptr, 0, nullptr, G0, 0, 512, ML, 512, 512);   // K
  gemm(T1, 512, WV1, 512, P[5], 1.f, 0, nullptr, 0, nullptr, G1, 0, 512, ML, 512, 512);   // V
  gemm(T1 + (L_ - 1) * D_, LD, WQ1, 512, P[1], sc, 0, nullptr, 0, S_Q1, nullptr, 0, 512, B_, 512, 512);
  attn_single_kernel<<<dim3(B_ * H_), dim3(128), 0, stream>>>(S_Q1, G0, G1, S_O1B);
  gemm(S_O1B, 512, WO1, 512, P[7], 1.f, 0, S_A1L, 512, S_Y1, S_Y1TB, 1, 512, B_, 512, 512); // y1 + tanh(y1)
  gemm(S_Y1TB, 512, WF1, 512, P[25], 1.f, 1, S_Y1, 512, S_T, S_TB, 1, 512, B_, 512, 512);   // o1 + tanh(o1)

  // ---- att2 (full) ----
  gemm(T2, 512, WQ2, 512, P[9],  sc,  0, nullptr, 0, nullptr, G2, 0, 512, ML, 512, 512);
  gemm(T2, 512, WK2, 512, P[11], 1.f, 0, nullptr, 0, nullptr, G0, 0, 512, ML, 512, 512);
  gemm(T2, 512, WV2, 512, P[13], 1.f, 0, nullptr, 0, nullptr, G1, 0, 512, ML, 512, 512);
  attn_full_kernel<<<dim3(B_ * H_), blk, 0, stream>>>(G2, G0, G1, T1);                 // O -> T1
  gemm(T1, 512, WO2, 512, P[15], 1.f, 0, FA2, 512, FY2, T2, 1, 512, ML, 512, 512);     // y2 + tanh(y2)
  gemm(T2, 512, WF2, 512, P[27], 1.f, 1, FY2, 512, nullptr, G2, 1, 512, ML, 512, 512); // tanh(o2)

  // ---- attx (query at L-1, kv = o2) ----
  gemm(G2, 512, WKX, 512, P[19], 1.f, 0, nullptr, 0, nullptr, G0, 0, 512, ML, 512, 512); // K
  gemm(G2, 512, WVX, 512, P[21], 1.f, 0, nullptr, 0, nullptr, G1, 0, 512, ML, 512, 512); // V
  gemm(S_TB, 512, WQX, 512, P[17], sc, 0, nullptr, 0, S_QX, nullptr, 0, 512, B_, 512, 512);
  attn_single_kernel<<<dim3(B_ * H_), dim3(128), 0, stream>>>(S_QX, G0, G1, S_OXB);
  gemm(S_OXB, 512, WOX, 512, P[23], 1.f, 0, S_T, 512, S_AX, S_AXTB, 1, 512, B_, 512, 512);

  // ---- GRU ----
  gemm(S_AXTB, 512, WI, 1536, P[29], 1.f, 0, nullptr, 0, S_GI, nullptr, 0, 1536, B_, 1536, 512);
  gemm(S_HB,   512, WH, 1536, nullptr, 1.f, 0, nullptr, 0, S_GH, nullptr, 0, 1536, B_, 1536, 512);
  gru_final_kernel<<<dim3(B_ * D_ / 256), blk, 0, stream>>>(S_GI, S_GH, S_H, S_AX,
                                                            P[31], out_h, out_y);
}
